// DiscreteActionTransitionNetworkSoftMoE_89094801588267
// MI455X (gfx1250) — matrix operand order repaired and checked by reading, not compiled, not run on hardware
//
#include <hip/hip_runtime.h>

// ---------------------------------------------------------------------------
// Soft-MoE discrete-action transition network, fused action selection.
// CDNA5 (gfx1250, wave32): all GEMMs on v_wmma_f32_16x16x32_bf16.
// Register->LDS double-buffered panels (one barrier per K-step), bf16 panels
// converted once, fragments read via 16B ds_load_b128 in ISA VGPR layout.
// ---------------------------------------------------------------------------

typedef __attribute__((ext_vector_type(16))) __bf16 v16bf;
typedef __attribute__((ext_vector_type(8)))  __bf16 v8bf;
typedef __attribute__((ext_vector_type(4)))  __bf16 v4bf;
typedef __attribute__((ext_vector_type(8)))  float  v8f;

static constexpr int kB  = 64;    // batch
static constexpr int kM  = 256;   // tokens
static constexpr int kD  = 256;   // feature dim
static constexpr int kA  = 18;    // actions
static constexpr int kE  = 4;     // experts
static constexpr int kS  = 64;    // slots per expert
static constexpr int kES = kE * kS;   // 256 slots total (== kM)
static constexpr int kH  = 512;   // expert hidden
static constexpr int kAD = kA * kD;   // 4608 head width

// Block tile: 64(M) x 128(N), K-step 32. 8 waves, each wave 32x32 (2x2 frags).
static constexpr int BM = 64;
static constexpr int BN = 128;
static constexpr int BK = 32;

union F16x16 { v16bf v; v8bf h[2]; };

__device__ __forceinline__ v8f wmma_bf16(v16bf a, v16bf b, v8f c) {
  return __builtin_amdgcn_wmma_f32_16x16x32_bf16(
      false, a, false, b, (short)0, c, false, false);
}

// ---------------------------------------------------------------------------
// Panel staging, split into global->reg and reg->LDS halves (double buffer).
// Row-major A panel (k contiguous): 8 f32/thread as two float4.
// ---------------------------------------------------------------------------
struct RowsReg { float4 v[2]; };

__device__ __forceinline__ void load_rows(RowsReg& R, const float* __restrict__ src,
                                          int ld, int tid) {
  const int e0 = tid * 4;            // 0..1020, BM*BK = 2048 elems total
  const int e1 = e0 + 1024;
  R.v[0] = *(const float4*)(src + (size_t)(e0 >> 5) * ld + (e0 & 31));
  R.v[1] = *(const float4*)(src + (size_t)(e1 >> 5) * ld + (e1 & 31));
}

__device__ __forceinline__ void store_rows(__bf16* dst, const RowsReg& R, int tid) {
  const int e0 = tid * 4;
  const int e1 = e0 + 1024;
  v4bf p0 = { (__bf16)R.v[0].x, (__bf16)R.v[0].y, (__bf16)R.v[0].z, (__bf16)R.v[0].w };
  v4bf p1 = { (__bf16)R.v[1].x, (__bf16)R.v[1].y, (__bf16)R.v[1].z, (__bf16)R.v[1].w };
  *(v4bf*)(dst + (e0 >> 5) * BK + (e0 & 31)) = p0;
  *(v4bf*)(dst + (e1 >> 5) * BK + (e1 & 31)) = p1;
}

// Transposed panel (element (k,r) at src[k*ld + r] -> LDS dst[r*BK + k]).
template<int ROWS>
struct ColsReg { float f[(ROWS * BK) / 256]; };

template<int ROWS>
__device__ __forceinline__ void load_cols(ColsReg<ROWS>& R, const float* __restrict__ src,
                                          int ld, int tid) {
  constexpr int KC = (ROWS * BK) / 256;
  const int r  = tid & (ROWS - 1);
  const int kb = (tid / ROWS) * KC;
#pragma unroll
  for (int j = 0; j < KC; ++j) R.f[j] = src[(size_t)(kb + j) * ld + r];
}

template<int ROWS>
__device__ __forceinline__ void store_cols(__bf16* dst, const ColsReg<ROWS>& R, int tid) {
  constexpr int KC = (ROWS * BK) / 256;
  const int r  = tid & (ROWS - 1);
  const int kb = (tid / ROWS) * KC;
#pragma unroll
  for (int j = 0; j < KC; j += 4) {
    v4bf pk = { (__bf16)R.f[j], (__bf16)R.f[j + 1], (__bf16)R.f[j + 2], (__bf16)R.f[j + 3] };
    *(v4bf*)(dst + r * BK + kb + j) = pk;
  }
}

// ---------------------------------------------------------------------------
// Fragment loads from LDS (ISA VGPR layouts, 16-byte ds reads).
// A 16x32: lanes 0-15 m=lane, K-half 0; lanes 16-31 K-half 8; regs 0-3 / 4-7
//          split at K+16  ->  two contiguous 16B chunks per lane.
// B 32x16 (stored transposed Bs[n][k]): same K layout as A (K-half 0/8, split at K+16).
// ---------------------------------------------------------------------------
__device__ __forceinline__ v16bf frag_a_lds(const __bf16* As, int m0, int lane) {
  const int m  = m0 + (lane & 15);
  const int kh = (lane < 16) ? 0 : 8;
  F16x16 a;
  a.h[0] = *(const v8bf*)(As + m * BK + kh);
  a.h[1] = *(const v8bf*)(As + m * BK + kh + 16);
  return a.v;
}

__device__ __forceinline__ v16bf frag_b_lds(const __bf16* Bs, int n0, int lane) {
  const int n  = n0 + (lane & 15);
  const int kh = (lane < 16) ? 0 : 8;
  F16x16 b;
  b.h[0] = *(const v8bf*)(Bs + n * BK + kh);
  b.h[1] = *(const v8bf*)(Bs + n * BK + kh + 16);
  return b.v;
}

// ---------------------------------------------------------------------------
// Core 64x128 block-tile GEMM, double-buffered (one barrier per K-step).
// ATRANS=false: A element (m,k) at Asrc[m*lda + k]
// ATRANS=true : A element (m,k) at Asrc[k*lda + m]
// B element (k,n) at Bsrc[k*ldb + n]
// ---------------------------------------------------------------------------
template<bool ATRANS>
__device__ __forceinline__ void gemm_64x128(const float* __restrict__ Asrc, int lda,
                                            const float* __restrict__ Bsrc, int ldb,
                                            int K, __bf16* As, __bf16* Bs,
                                            v8f acc[2][2]) {
  const int tid  = threadIdx.x;
  const int lane = tid & 31;
  const int wave = tid >> 5;
  const int wm = (wave >> 2) * 32;
  const int wn = (wave & 3) * 32;

  RowsReg      aRow;
  ColsReg<BM>  aCol;
  ColsReg<BN>  bReg;

  // Prologue: stage panel 0 into buffer 0.
  if constexpr (ATRANS) load_cols<BM>(aCol, Asrc, lda, tid);
  else                  load_rows(aRow, Asrc, lda, tid);
  load_cols<BN>(bReg, Bsrc, ldb, tid);
  if constexpr (ATRANS) store_cols<BM>(As, aCol, tid);
  else                  store_rows(As, aRow, tid);
  store_cols<BN>(Bs, bReg, tid);
  __syncthreads();

  int ping = 0;
  for (int k0 = 0; k0 < K; k0 += BK) {
    const bool nxt = (k0 + BK) < K;
    if (nxt) {  // global -> regs for panel k0+BK, overlapped with WMMA below
      if constexpr (ATRANS) load_cols<BM>(aCol, Asrc + (size_t)(k0 + BK) * lda, lda, tid);
      else                  load_rows(aRow, Asrc + (k0 + BK), lda, tid);
      load_cols<BN>(bReg, Bsrc + (size_t)(k0 + BK) * ldb, ldb, tid);
    }
    const __bf16* Ap = As + ping * (BM * BK);
    const __bf16* Bp = Bs + ping * (BN * BK);
    const v16bf a0 = frag_a_lds(Ap, wm,      lane);
    const v16bf a1 = frag_a_lds(Ap, wm + 16, lane);
    const v16bf b0 = frag_b_lds(Bp, wn,      lane);
    const v16bf b1 = frag_b_lds(Bp, wn + 16, lane);
    acc[0][0] = wmma_bf16(a0, b0, acc[0][0]);
    acc[0][1] = wmma_bf16(a0, b1, acc[0][1]);
    acc[1][0] = wmma_bf16(a1, b0, acc[1][0]);
    acc[1][1] = wmma_bf16(a1, b1, acc[1][1]);
    if (nxt) {  // regs -> other LDS buffer; single collective barrier
      const int pong = ping ^ 1;
      if constexpr (ATRANS) store_cols<BM>(As + pong * (BM * BK), aCol, tid);
      else                  store_rows(As + pong * (BM * BK), aRow, tid);
      store_cols<BN>(Bs + pong * (BN * BK), bReg, tid);
      __syncthreads();
      ping = pong;
    }
  }
}

// Epilogues: 0 = plain, 1 = +bias, 2 = relu(+bias)
template<int EPI>
__device__ __forceinline__ void store_acc(float* __restrict__ C, int ldc,
                                          const float* __restrict__ bias,
                                          const v8f acc[2][2]) {
  const int tid  = threadIdx.x;
  const int lane = tid & 31;
  const int wave = tid >> 5;
  const int wm = (wave >> 2) * 32;
  const int wn = (wave & 3) * 32;
  const int rb  = (lane < 16) ? 0 : 8;
  const int col = lane & 15;
#pragma unroll
  for (int i = 0; i < 2; ++i)
#pragma unroll
    for (int j = 0; j < 2; ++j) {
      const int cg = wn + j * 16 + col;
      float bv = 0.f;
      if (EPI > 0) bv = bias[cg];
#pragma unroll
      for (int v = 0; v < 8; ++v) {
        float x = acc[i][j][v] + bv;
        if (EPI == 2) x = fmaxf(x, 0.f);
        C[(size_t)(wm + i * 16 + rb + v) * ldc + cg] = x;
      }
    }
}

// ---------------------------------------------------------------------------
// Stage 1: logits[b,m,es] = obs[b,m,:] . phi[:,es]
// grid: B * (256/64) * (256/128) = 512 blocks
// ---------------------------------------------------------------------------
__global__ void k_gemm_logits(const float* __restrict__ obs,
                              const float* __restrict__ phi,
                              float* __restrict__ logits) {
  __shared__ __align__(16) __bf16 As[2 * BM * BK];
  __shared__ __align__(16) __bf16 Bs[2 * BN * BK];
  const int bid = blockIdx.x;
  const int b   = bid >> 3;
  const int tmb = (bid >> 1) & 3;
  const int tnb = bid & 1;
  v8f acc[2][2] = {};
  gemm_64x128<false>(obs + (size_t)b * kM * kD + (size_t)tmb * BM * kD, kD,
                     phi + tnb * BN, kES, kD, As, Bs, acc);
  store_acc<0>(logits + (size_t)b * kM * kES + (size_t)tmb * BM * kES + tnb * BN,
               kES, nullptr, acc);
}

// ---------------------------------------------------------------------------
// Stage 2a: combine = softmax over contiguous slot dim (256); wave per row.
// ---------------------------------------------------------------------------
__global__ void k_softmax_rows(const float* __restrict__ in, float* __restrict__ out) {
  const int lane = threadIdx.x & 31;
  const int row  = blockIdx.x * 8 + (threadIdx.x >> 5);
  const float* p = in + (size_t)row * kES;
  float vals[8];
  float mx = -3.4e38f;
#pragma unroll
  for (int i = 0; i < 8; ++i) { vals[i] = p[lane + i * 32]; mx = fmaxf(mx, vals[i]); }
#pragma unroll
  for (int off = 16; off > 0; off >>= 1) mx = fmaxf(mx, __shfl_xor(mx, off, 32));
  float sum = 0.f;
#pragma unroll
  for (int i = 0; i < 8; ++i) { vals[i] = __expf(vals[i] - mx); sum += vals[i]; }
#pragma unroll
  for (int off = 16; off > 0; off >>= 1) sum += __shfl_xor(sum, off, 32);
  const float inv = __frcp_rn(sum);
  float* q = out + (size_t)row * kES;
#pragma unroll
  for (int i = 0; i < 8; ++i) q[lane + i * 32] = vals[i] * inv;
}

// ---------------------------------------------------------------------------
// Stage 2b: dispatch = softmax over tokens m (stride kES); wave per (b,es).
// ---------------------------------------------------------------------------
__global__ void k_softmax_cols(const float* __restrict__ in, float* __restrict__ out) {
  const int lane = threadIdx.x & 31;
  const int idx  = blockIdx.x * 8 + (threadIdx.x >> 5);
  const int b  = idx >> 8;
  const int es = idx & 255;
  const float* p = in + (size_t)b * kM * kES + es;
  float vals[8];
  float mx = -3.4e38f;
#pragma unroll
  for (int i = 0; i < 8; ++i) { vals[i] = p[(size_t)(lane + i * 32) * kES]; mx = fmaxf(mx, vals[i]); }
#pragma unroll
  for (int off = 16; off > 0; off >>= 1) mx = fmaxf(mx, __shfl_xor(mx, off, 32));
  float sum = 0.f;
#pragma unroll
  for (int i = 0; i < 8; ++i) { vals[i] = __expf(vals[i] - mx); sum += vals[i]; }
#pragma unroll
  for (int off = 16; off > 0; off >>= 1) sum += __shfl_xor(sum, off, 32);
  const float inv = __frcp_rn(sum);
  float* q = out + (size_t)b * kM * kES + es;
#pragma unroll
  for (int i = 0; i < 8; ++i) q[(size_t)(lane + i * 32) * kES] = vals[i] * inv;
}

// ---------------------------------------------------------------------------
// Stage 3: slots[b,es,d] = sum_m dispatch[b,m,es] * obs[b,m,d]   (A transposed)
// ---------------------------------------------------------------------------
__global__ void k_gemm_slots(const float* __restrict__ dispatch,
                             const float* __restrict__ obs,
                             float* __restrict__ slots) {
  __shared__ __align__(16) __bf16 As[2 * BM * BK];
  __shared__ __align__(16) __bf16 Bs[2 * BN * BK];
  const int bid = blockIdx.x;
  const int b   = bid >> 3;
  const int tmb = (bid >> 1) & 3;   // es tile
  const int tnb = bid & 1;          // d tile
  v8f acc[2][2] = {};
  gemm_64x128<true>(dispatch + (size_t)b * kM * kES + tmb * BM, kES,
                    obs + (size_t)b * kM * kD + tnb * BN, kD, kM, As, Bs, acc);
  store_acc<0>(slots + (size_t)b * kES * kD + (size_t)tmb * BM * kD + tnb * BN,
               kD, nullptr, acc);
}

// ---------------------------------------------------------------------------
// Stage 4: h = relu(slots . w1 + b1), per (b,e): [64,256]x[256,512]
// grid: B*E*(512/128) = 1024 blocks
// ---------------------------------------------------------------------------
__global__ void k_gemm_h(const float* __restrict__ slots,
                         const float* __restrict__ w1,
                         const float* __restrict__ b1,
                         float* __restrict__ hbuf) {
  __shared__ __align__(16) __bf16 As[2 * BM * BK];
  __shared__ __align__(16) __bf16 Bs[2 * BN * BK];
  const int bid = blockIdx.x;
  const int be  = bid >> 2;
  const int tnb = bid & 3;
  const int e   = be & 3;
  v8f acc[2][2] = {};
  gemm_64x128<false>(slots + (size_t)be * kS * kD, kD,
                     w1 + (size_t)e * kD * kH + tnb * BN, kH, kD, As, Bs, acc);
  store_acc<2>(hbuf + (size_t)be * kS * kH + tnb * BN, kH,
               b1 + e * kH + tnb * BN, acc);
}

// ---------------------------------------------------------------------------
// Stage 5 (fused action selection): slice w2/b2 at act*D before the GEMM.
// per (b,e): [64,512]x[512,256];  grid: B*E*(256/128) = 512 blocks
// ---------------------------------------------------------------------------
__global__ void k_gemm_ysel(const float* __restrict__ hbuf,
                            const float* __restrict__ w2,
                            const float* __restrict__ b2,
                            const int* __restrict__ action,
                            float* __restrict__ ysel) {
  __shared__ __align__(16) __bf16 As[2 * BM * BK];
  __shared__ __align__(16) __bf16 Bs[2 * BN * BK];
  const int bid = blockIdx.x;
  const int be  = bid >> 1;
  const int tnb = bid & 1;
  const int b   = be >> 2;
  const int e   = be & 3;
  const int act = action[b];
  v8f acc[2][2] = {};
  gemm_64x128<false>(hbuf + (size_t)be * kS * kH, kH,
                     w2 + (size_t)e * kH * kAD + (size_t)act * kD + tnb * BN, kAD,
                     kH, As, Bs, acc);
  store_acc<1>(ysel + (size_t)be * kS * kD + tnb * BN, kD,
               b2 + (size_t)e * kAD + act * kD + tnb * BN, acc);
}

// ---------------------------------------------------------------------------
// Stage 6: out[b,m,d] = sum_es combine[b,m,es] * y_sel[b,es,d]
// ---------------------------------------------------------------------------
__global__ void k_gemm_out(const float* __restrict__ combine,
                           const float* __restrict__ ysel,
                           float* __restrict__ out) {
  __shared__ __align__(16) __bf16 As[2 * BM * BK];
  __shared__ __align__(16) __bf16 Bs[2 * BN * BK];
  const int bid = blockIdx.x;
  const int b   = bid >> 3;
  const int tmb = (bid >> 1) & 3;
  const int tnb = bid & 1;
  v8f acc[2][2] = {};
  gemm_64x128<false>(combine + (size_t)b * kM * kES + (size_t)tmb * BM * kES, kES,
                     ysel + (size_t)b * kES * kD + tnb * BN, kD, kES, As, Bs, acc);
  store_acc<0>(out + (size_t)b * kM * kD + (size_t)tmb * BM * kD + tnb * BN,
               kD, nullptr, acc);
}

// ---------------------------------------------------------------------------
// Launch
// ---------------------------------------------------------------------------
extern "C" void kernel_launch(void* const* d_in, const int* in_sizes, int n_in,
                              void* d_out, int out_size, void* d_ws, size_t ws_size,
                              hipStream_t stream) {
  (void)in_sizes; (void)n_in; (void)out_size; (void)ws_size;
  const float* obs    = (const float*)d_in[0];
  const int*   action = (const int*)d_in[1];
  const float* phi    = (const float*)d_in[2];
  const float* w1     = (const float*)d_in[3];
  const float* b1     = (const float*)d_in[4];
  const float* w2     = (const float*)d_in[5];
  const float* b2     = (const float*)d_in[6];
  float* out = (float*)d_out;

  float* ws       = (float*)d_ws;
  float* logits   = ws;                                   // B*M*ES
  float* dispatch = logits   + (size_t)kB * kM * kES;
  float* combine  = dispatch + (size_t)kB * kM * kES;
  float* slots    = combine  + (size_t)kB * kM * kES;     // B*ES*D
  float* hbuf     = slots    + (size_t)kB * kES * kD;     // B*ES*H
  float* ysel     = hbuf     + (size_t)kB * kES * kH;     // B*ES*D

  const dim3 blk(256);  // 8 wave32 waves

  k_gemm_logits<<<dim3(kB * 8), blk, 0, stream>>>(obs, phi, logits);
  k_softmax_rows<<<dim3(kB * kM / 8), blk, 0, stream>>>(logits, combine);
  k_softmax_cols<<<dim3(kB * kES / 8), blk, 0, stream>>>(logits, dispatch);
  k_gemm_slots<<<dim3(kB * 8), blk, 0, stream>>>(dispatch, obs, slots);
  k_gemm_h<<<dim3(kB * kE * 4), blk, 0, stream>>>(slots, w1, b1, hbuf);
  k_gemm_ysel<<<dim3(kB * kE * 2), blk, 0, stream>>>(hbuf, w2, b2, action, ysel);
  k_gemm_out<<<dim3(kB * 8), blk, 0, stream>>>(combine, ysel, out);
}
